// MultiHeadAttentionQuantum_65481071397626
// MI455X (gfx1250) — matrix operand order repaired and checked by reading, not compiled, not run on hardware
//
#include <hip/hip_runtime.h>
#include <hip/hip_bf16.h>

typedef __attribute__((ext_vector_type(16))) _Float16 v16h;
typedef __attribute__((ext_vector_type(8)))  _Float16 v8h;
typedef __attribute__((ext_vector_type(4)))  _Float16 v4h;
typedef __attribute__((ext_vector_type(8)))  float    v8f;
typedef __attribute__((ext_vector_type(4)))  float    v4f;

static constexpr int   kE         = 1024;   // E = H*D = N = K
static constexpr int   kMTile     = 64;     // rows per block (4 row-tiles of 16)
static constexpr int   kLdsStride = 1032;   // halfs; 1032%8==0 (16B rows), 516 dw -> 4-bank skew/row
static constexpr float kInv2Pi    = 0.15915494309189535f; // v_cos_f32 wants radians * 1/(2*pi)

// ---------------- Pre-pass: convert f32 weights -> f16 in workspace ----------------
__global__ __launch_bounds__(256)
void convert_weights_f16(const float* __restrict__ Wp, const float* __restrict__ Wc,
                         _Float16* __restrict__ wsh) {
  const size_t i4 = ((size_t)blockIdx.x * 256 + threadIdx.x) * 4;  // grid covers E*E elements
  const size_t n  = (size_t)kE * kE;
  {
    v4f w = *(const v4f*)(Wp + i4);
    v4h h; h.x = (_Float16)w.x; h.y = (_Float16)w.y; h.z = (_Float16)w.z; h.w = (_Float16)w.w;
    *(v4h*)(wsh + i4) = h;
  }
  {
    v4f w = *(const v4f*)(Wc + i4);
    v4h h; h.x = (_Float16)w.x; h.y = (_Float16)w.y; h.z = (_Float16)w.z; h.w = (_Float16)w.w;
    *(v4h*)(wsh + n + i4) = h;
  }
}

// A-fragment (16x32 f16) from LDS, row-major tile with stride kLdsStride.
// ISA layout: lanes 0-15 = rows M=0..15 holding K = kg..kg+7 and kg+16..kg+23 (kg = 8*(lane>=16)).
__device__ __forceinline__ v16h load_a_frag_lds(const _Float16* lds, int rbase, int kb, int lane) {
  const int row = rbase + (lane & 15);
  const int kg  = (lane >> 4) * 8;
  const _Float16* p = lds + row * kLdsStride + kb + kg;
  v8h lo = *(const v8h*)(p);        // ds_load_b128
  v8h hi = *(const v8h*)(p + 16);   // ds_load_b128
  return __builtin_shufflevector(lo, hi, 0, 1, 2, 3, 4, 5, 6, 7,
                                         8, 9, 10, 11, 12, 13, 14, 15);
}

// B-fragment (32x16 f16) from pre-converted row-major [N,K] f16 weight.
// ISA layout: lane%16 = column n, holding K = kg..kg+7 and kg+16..kg+23 (kg = 8*(lane>=16)).
__device__ __forceinline__ v16h load_b_frag_h(const _Float16* __restrict__ Wh, int cbase, int kb, int lane) {
  const _Float16* p = Wh + (size_t)(cbase + (lane & 15)) * kE + kb + (lane >> 4) * 8;
  v8h lo = *(const v8h*)(p);        // global_load_b128
  v8h hi = *(const v8h*)(p + 16);   // global_load_b128
  return __builtin_shufflevector(lo, hi, 0, 1, 2, 3, 4, 5, 6, 7,
                                         8, 9, 10, 11, 12, 13, 14, 15);
}

__global__ __launch_bounds__(256, 1)
void MultiHeadAttentionQuantum_fused(const float* __restrict__ x,
                                     const _Float16* __restrict__ Wph,
                                     const _Float16* __restrict__ Wch,
                                     const float* __restrict__ phi,
                                     float* __restrict__ out) {
  __shared__ _Float16 Xs[kMTile * kLdsStride];  // x tile, f16          (~132 KB)
  __shared__ _Float16 Qs[kMTile * kLdsStride];  // cos(proj+phi), f16   (~132 KB)
  __shared__ float    Ph[kE];                   // flat phi

  const int    tid   = threadIdx.x;
  const int    lane  = tid & 31;
  const int    wave  = tid >> 5;                // 8 wave32s
  const size_t mbase = (size_t)blockIdx.x * kMTile;

  for (int i = tid; i < kE; i += 256) Ph[i] = phi[i];

  // ---- Stage X tile: global f32 -> LDS f16 (each thread: 1/4 of one row) ----
  {
    const int row = tid >> 2;                   // 0..63
    const int c0  = (tid & 3) * (kE / 4);       // 0,256,512,768
    const float* src = x + (mbase + row) * kE + c0;
    _Float16*    dst = &Xs[row * kLdsStride + c0];
#pragma unroll 8
    for (int j = 0; j < kE / 4; j += 4) {
      v4f v = *(const v4f*)(src + j);
      v4h h;
      h.x = (_Float16)v.x; h.y = (_Float16)v.y;
      h.z = (_Float16)v.z; h.w = (_Float16)v.w;
      *(v4h*)(dst + j) = h;                     // ds_store_b64
    }
  }
  __syncthreads();

  const int nCol = kE / 16;                     // 64 col-tiles; each wave: 1 col-tile x 4 row-tiles

  // ---- Phase A: proj = Xs @ Wp^T ; Qs = f16(cos(proj + phi)) ----
  for (int c = wave; c < nCol; c += 8) {
    // Stop LICM from hoisting (and spilling) the whole K-sweep of LDS A-fragments
    // out of this loop: force per-iteration ds_loads.
    asm volatile("" ::: "memory");
    const int cbase = c * 16;
    v8f acc0 = {}, acc1 = {}, acc2 = {}, acc3 = {};
#pragma unroll 2
    for (int kb = 0; kb < kE; kb += 32) {
      v16h b  = load_b_frag_h(Wph, cbase, kb, lane);   // shared by 4 row-tiles
      if (kb + 32 < kE)                                // global_prefetch_b8 of next fragment
        __builtin_prefetch(Wph + (size_t)(cbase + (lane & 15)) * kE + (kb + 32) + (lane >> 4) * 16, 0, 3);
      v16h a0 = load_a_frag_lds(Xs, 0,  kb, lane);
      v16h a1 = load_a_frag_lds(Xs, 16, kb, lane);
      v16h a2 = load_a_frag_lds(Xs, 32, kb, lane);
      v16h a3 = load_a_frag_lds(Xs, 48, kb, lane);
      acc0 = __builtin_amdgcn_wmma_f32_16x16x32_f16(false, a0, false, b, (short)0, acc0, false, false);
      acc1 = __builtin_amdgcn_wmma_f32_16x16x32_f16(false, a1, false, b, (short)0, acc1, false, false);
      acc2 = __builtin_amdgcn_wmma_f32_16x16x32_f16(false, a2, false, b, (short)0, acc2, false, false);
      acc3 = __builtin_amdgcn_wmma_f32_16x16x32_f16(false, a3, false, b, (short)0, acc3, false, false);
    }
    // C/D layout: VGPR v -> M = v + 8*(lane>=16); lane%16 -> N
    const int   n  = lane & 15;
    const int   mh = (lane >> 4) * 8;
    const float ph = Ph[cbase + n];
    _Float16* q = &Qs[(mh)*kLdsStride + cbase + n];
#pragma unroll
    for (int v = 0; v < 8; ++v) {
      q[(v)      * kLdsStride] = (_Float16)__builtin_amdgcn_cosf((acc0[v] + ph) * kInv2Pi);
      q[(16 + v) * kLdsStride] = (_Float16)__builtin_amdgcn_cosf((acc1[v] + ph) * kInv2Pi);
      q[(32 + v) * kLdsStride] = (_Float16)__builtin_amdgcn_cosf((acc2[v] + ph) * kInv2Pi);
      q[(48 + v) * kLdsStride] = (_Float16)__builtin_amdgcn_cosf((acc3[v] + ph) * kInv2Pi);
    }
  }
  __syncthreads();

  // ---- Phase B: out = Qs @ Wc^T (f32 out) ----
  for (int c = wave; c < nCol; c += 8) {
    asm volatile("" ::: "memory");
    const int cbase = c * 16;
    v8f acc0 = {}, acc1 = {}, acc2 = {}, acc3 = {};
#pragma unroll 2
    for (int kb = 0; kb < kE; kb += 32) {
      v16h b  = load_b_frag_h(Wch, cbase, kb, lane);
      if (kb + 32 < kE)
        __builtin_prefetch(Wch + (size_t)(cbase + (lane & 15)) * kE + (kb + 32) + (lane >> 4) * 16, 0, 3);
      v16h a0 = load_a_frag_lds(Qs, 0,  kb, lane);
      v16h a1 = load_a_frag_lds(Qs, 16, kb, lane);
      v16h a2 = load_a_frag_lds(Qs, 32, kb, lane);
      v16h a3 = load_a_frag_lds(Qs, 48, kb, lane);
      acc0 = __builtin_amdgcn_wmma_f32_16x16x32_f16(false, a0, false, b, (short)0, acc0, false, false);
      acc1 = __builtin_amdgcn_wmma_f32_16x16x32_f16(false, a1, false, b, (short)0, acc1, false, false);
      acc2 = __builtin_amdgcn_wmma_f32_16x16x32_f16(false, a2, false, b, (short)0, acc2, false, false);
      acc3 = __builtin_amdgcn_wmma_f32_16x16x32_f16(false, a3, false, b, (short)0, acc3, false, false);
    }
    const int n  = lane & 15;
    const int mh = (lane >> 4) * 8;
    float* op = out + (mbase + mh) * kE + cbase + n;
#pragma unroll
    for (int v = 0; v < 8; ++v) {
      op[(size_t)(v)      * kE] = acc0[v];
      op[(size_t)(16 + v) * kE] = acc1[v];
      op[(size_t)(32 + v) * kE] = acc2[v];
      op[(size_t)(48 + v) * kE] = acc3[v];
    }
  }
}

extern "C" void kernel_launch(void* const* d_in, const int* in_sizes, int n_in,
                              void* d_out, int out_size, void* d_ws, size_t ws_size,
                              hipStream_t stream) {
  (void)n_in; (void)ws_size; (void)out_size;
  const float* x   = (const float*)d_in[0];   // [B,S,E] f32
  const float* Wp  = (const float*)d_in[1];   // [E,E]   f32
  const float* Wc  = (const float*)d_in[2];   // [E,E]   f32
  const float* phi = (const float*)d_in[3];   // [H,D] -> flat [E] f32
  float*       out = (float*)d_out;           // [B,S,E] f32

  _Float16* wsh = (_Float16*)d_ws;            // 2 * E*E f16 = 4 MB scratch
  const size_t n = (size_t)kE * kE;

  // Pre-pass: convert both weight matrices to f16 once.
  {
    dim3 grid((unsigned)(n / (256 * 4)));
    hipLaunchKernelGGL(convert_weights_f16, grid, dim3(256), 0, stream, Wp, Wc, wsh);
  }

  // Fused GEMM -> cos -> GEMM.
  const int M = in_sizes[0] / kE;             // B*S = 32768
  dim3 grid(M / kMTile);
  hipLaunchKernelGGL(MultiHeadAttentionQuantum_fused, grid, dim3(256), 0, stream,
                     x, wsh, wsh + n, phi, out);
}
